// RobertaSoftUnifSelfAttention_24455543783776
// MI455X (gfx1250) — matrix operand order repaired and checked by reading, compiled, not run on hardware
//
#include <hip/hip_runtime.h>

#define S_LEN 2048
#define HIDN  1024
#define NH    16
#define DHD   64
#define TDD   32

typedef __attribute__((ext_vector_type(16))) __bf16 v16bf;
typedef __attribute__((ext_vector_type(8)))  float  v8f;

// ---------------------------------------------------------------------------
// Fragment helpers (wave32; k order contiguous in A and B, not the ISA table order):
//  A 16x32 bf16 : lane l -> row M = l&15 ; kg = l>>4 ;
//                 e<8 : K = kg*16+e       (16B contiguous)
//                 e>=8: K = kg*16+e       (next 16B, contiguous)
//  B 32x16 bf16 : lane l -> col N = l&15 ; K = (l>>4)*16 + e  (32B contiguous)
//  C/D 16x16 f32: lane l, vgpr r -> M = (l>>4)*8 + r, N = l&15
// ---------------------------------------------------------------------------
__device__ __forceinline__ v16bf load_frag2(const __bf16* p0, const __bf16* p1) {
    union { v16bf v; uint4 q[2]; } u;
    u.q[0] = *reinterpret_cast<const uint4*>(p0);
    u.q[1] = *reinterpret_cast<const uint4*>(p1);
    return u.v;
}

__device__ __forceinline__ v8f wmma_bf16(v16bf a, v16bf b, v8f c) {
    return __builtin_amdgcn_wmma_f32_16x16x32_bf16(false, a, false, b,
                                                   (short)0, c, false, false);
}

// ---------------------------------------------------------------------------
// f32 -> bf16 convert (x activations)
// ---------------------------------------------------------------------------
__global__ void cvt_f32_bf16(const float* __restrict__ in,
                             __bf16* __restrict__ out, int n) {
    int i = blockIdx.x * blockDim.x + threadIdx.x;
    if (i < n) out[i] = (__bf16)in[i];
}

// W[K][N] f32 row-major  ->  Wt[N][K] bf16 (so B-fragments are contiguous)
__global__ void transpose_cvt(const float* __restrict__ W,
                              __bf16* __restrict__ Wt, int K, int N) {
    int i = blockIdx.x * blockDim.x + threadIdx.x;
    if (i < K * N) {
        int k = i / N, n = i % N;
        Wt[(size_t)n * K + k] = (__bf16)W[i];
    }
}

// ---------------------------------------------------------------------------
// GEMM: C[M x N] = A[M x K](bf16) * B (given as Bt[N x K] bf16) + bias[N]
// block = 128 thr = 4 waves; block tile 32(M) x 64(N); wave tile 16 x 32.
// ---------------------------------------------------------------------------
__global__ void gemm_bf16(const __bf16* __restrict__ A,
                          const __bf16* __restrict__ Bt,
                          const float*  __restrict__ bias,
                          float* __restrict__ C, int M, int N, int K) {
    const int w    = threadIdx.x >> 5;
    const int lane = threadIdx.x & 31;
    const int kg   = lane >> 4;
    const int ln   = lane & 15;

    const int mrow = blockIdx.y * 32 + (w & 1) * 16;
    const int ncol = blockIdx.x * 64 + (w >> 1) * 32;

    const __bf16* ap  = A  + (size_t)(mrow + ln) * K;
    const __bf16* bp0 = Bt + (size_t)(ncol + ln) * K + kg * 16;
    const __bf16* bp1 = Bt + (size_t)(ncol + 16 + ln) * K + kg * 16;

    v8f acc0 = {}, acc1 = {};
    for (int k0 = 0; k0 < K; k0 += 32) {
        __builtin_prefetch(ap  + k0 + 256, 0, 1);
        __builtin_prefetch(bp0 + k0 + 256, 0, 1);
        v16bf a  = load_frag2(ap + k0 + kg * 16, ap + k0 + kg * 16 + 8);
        v16bf b0 = load_frag2(bp0 + k0, bp0 + k0 + 8);
        v16bf b1 = load_frag2(bp1 + k0, bp1 + k0 + 8);
        acc0 = wmma_bf16(a, b0, acc0);
        acc1 = wmma_bf16(a, b1, acc1);
    }
#pragma unroll
    for (int r = 0; r < 8; ++r) {
        int row = mrow + kg * 8 + r;
        C[(size_t)row * N + ncol + ln]      = acc0[r] + bias[ncol + ln];
        C[(size_t)row * N + ncol + 16 + ln] = acc1[r] + bias[ncol + 16 + ln];
    }
}

// ---------------------------------------------------------------------------
// Prep: RoPE(Q,K) -> bf16 [H][S][64]; V -> bf16 transposed [H][64][S];
// KT -> bf16 [H][S][32]; QTW = qt @ W_bil[h] -> bf16 [H][S][32].
// grid (S, H), 64 threads (thread = dim d).
// ---------------------------------------------------------------------------
__global__ void prep_kernel(const float* __restrict__ Qf, const float* __restrict__ Kf,
                            const float* __restrict__ Vf, const float* __restrict__ QTf,
                            const float* __restrict__ KTf, const float* __restrict__ Wbil,
                            __bf16* __restrict__ Qr, __bf16* __restrict__ Kr,
                            __bf16* __restrict__ Vt, __bf16* __restrict__ QTW,
                            __bf16* __restrict__ KTo) {
    const int s = blockIdx.x, h = blockIdx.y, d = threadIdx.x;
    const int j = d & 31;
    // inv_freq = 10000^(-j/32)
    const float inv_freq = __expf(-9.210340371976184f * (float)j * (1.0f / 32.0f));
    const float fr = (float)s * inv_freq;
    const float c = __cosf(fr), sn = __sinf(fr);

    const size_t base = (size_t)s * HIDN + h * DHD;
    const float q  = Qf[base + d];
    const float k  = Kf[base + d];
    const float qo = (d < 32) ? Qf[base + d + 32] : Qf[base + d - 32];
    const float ko = (d < 32) ? Kf[base + d + 32] : Kf[base + d - 32];
    const float rq = (d < 32) ? -qo : qo;
    const float rk = (d < 32) ? -ko : ko;

    const size_t hs = (size_t)h * S_LEN + s;
    Qr[hs * DHD + d] = (__bf16)(q * c + rq * sn);
    Kr[hs * DHD + d] = (__bf16)(k * c + rk * sn);
    Vt[((size_t)h * DHD + d) * S_LEN + s] = (__bf16)Vf[base + d];

    if (d < TDD) {
        const size_t tbase = (size_t)s * (NH * TDD) + h * TDD;
        KTo[hs * TDD + d] = (__bf16)KTf[tbase + d];
        float acc = 0.0f;
#pragma unroll
        for (int t = 0; t < TDD; ++t)
            acc += QTf[tbase + t] * Wbil[((size_t)h * TDD + t) * TDD + d];
        QTW[hs * TDD + d] = (__bf16)acc;
    }
}

// ---------------------------------------------------------------------------
// Flash-style attention. grid (S/64, H), 128 thr = 4 waves.
// Each wave: 16 query rows, loops over keys in 32-wide tiles.
// Per tile: 4 WMMA (QK^T, K=64) + 2 WMMA (type score, K=32) + 4 WMMA (P*V).
// P transposed C-layout -> A-layout through padded LDS tile.
// ---------------------------------------------------------------------------
__global__ void attn_kernel(const __bf16* __restrict__ Qr, const __bf16* __restrict__ Kr,
                            const __bf16* __restrict__ Vt, const __bf16* __restrict__ QTW,
                            const __bf16* __restrict__ KTb, const float* __restrict__ tbias,
                            float* __restrict__ out) {
    const int w    = threadIdx.x >> 5;
    const int lane = threadIdx.x & 31;
    const int kg   = lane >> 4;
    const int ln   = lane & 15;
    const int h    = blockIdx.y;
    const int qbase = blockIdx.x * 64 + w * 16;

    const __bf16* Qh  = Qr  + (size_t)h * S_LEN * DHD;
    const __bf16* Kh  = Kr  + (size_t)h * S_LEN * DHD;
    const __bf16* Vh  = Vt  + (size_t)h * DHD * S_LEN;
    const __bf16* QTh = QTW + (size_t)h * S_LEN * TDD;
    const __bf16* KTh = KTb + (size_t)h * S_LEN * TDD;
    const float tb = tbias[h];

    // Persistent A-fragments for this wave's 16 query rows
    const int m = qbase + ln;
    const v16bf qa0 = load_frag2(Qh + (size_t)m * DHD + kg * 16,
                                 Qh + (size_t)m * DHD + kg * 16 + 8);
    const v16bf qa1 = load_frag2(Qh + (size_t)m * DHD + 32 + kg * 16,
                                 Qh + (size_t)m * DHD + 32 + kg * 16 + 8);
    const v16bf qta = load_frag2(QTh + (size_t)m * TDD + kg * 16,
                                 QTh + (size_t)m * TDD + kg * 16 + 8);

    float mrow[8], lrow[8];
#pragma unroll
    for (int r = 0; r < 8; ++r) { mrow[r] = -1e30f; lrow[r] = 0.0f; }
    v8f o0 = {}, o1 = {}, o2 = {}, o3 = {};

    __shared__ float lds[4][16][33];   // padded P tile per wave
    float (*P)[33] = lds[w];

    for (int kb = 0; kb < S_LEN; kb += 32) {
        // ---- score WMMAs ----
        v8f z = {};
        const __bf16* kp0 = Kh + (size_t)(kb + ln) * DHD + kg * 16;
        const __bf16* kp1 = Kh + (size_t)(kb + 16 + ln) * DHD + kg * 16;
        v8f sv0 = wmma_bf16(qa0, load_frag2(kp0, kp0 + 8), z);
        sv0     = wmma_bf16(qa1, load_frag2(kp0 + 32, kp0 + 40), sv0);
        v8f sv1 = wmma_bf16(qa0, load_frag2(kp1, kp1 + 8), z);
        sv1     = wmma_bf16(qa1, load_frag2(kp1 + 32, kp1 + 40), sv1);
        const __bf16* tp0 = KTh + (size_t)(kb + ln) * TDD + kg * 16;
        const __bf16* tp1 = KTh + (size_t)(kb + 16 + ln) * TDD + kg * 16;
        v8f st0 = wmma_bf16(qta, load_frag2(tp0, tp0 + 8), z);
        v8f st1 = wmma_bf16(qta, load_frag2(tp1, tp1 + 8), z);

        // ---- scores = qk/8 + log(sigmoid(type)+eps) ----
        float s0[8], s1[8];
#pragma unroll
        for (int r = 0; r < 8; ++r) {
            float t0 = st0[r] + tb;
            float t1 = st1[r] + tb;
            float g0 = 1.0f / (1.0f + __expf(-t0));
            float g1 = 1.0f / (1.0f + __expf(-t1));
            s0[r] = sv0[r] * 0.125f + __logf(g0 + 1e-6f);
            s1[r] = sv1[r] * 0.125f + __logf(g1 + 1e-6f);
        }

        // ---- online softmax update (row reductions stay inside 16-lane half) ----
        float p0[8], p1[8], alpha[8];
#pragma unroll
        for (int r = 0; r < 8; ++r) {
            float v = fmaxf(s0[r], s1[r]);
            v = fmaxf(v, __shfl_xor(v, 1, 32));
            v = fmaxf(v, __shfl_xor(v, 2, 32));
            v = fmaxf(v, __shfl_xor(v, 4, 32));
            v = fmaxf(v, __shfl_xor(v, 8, 32));
            float mnew = fmaxf(mrow[r], v);
            alpha[r] = __expf(mrow[r] - mnew);
            mrow[r]  = mnew;
            p0[r] = __expf(s0[r] - mnew);
            p1[r] = __expf(s1[r] - mnew);
            float rs = p0[r] + p1[r];
            rs += __shfl_xor(rs, 1, 32);
            rs += __shfl_xor(rs, 2, 32);
            rs += __shfl_xor(rs, 4, 32);
            rs += __shfl_xor(rs, 8, 32);
            lrow[r] = lrow[r] * alpha[r] + rs;
        }
#pragma unroll
        for (int r = 0; r < 8; ++r) {
            o0[r] *= alpha[r]; o1[r] *= alpha[r];
            o2[r] *= alpha[r]; o3[r] *= alpha[r];
        }

        // ---- transpose P (C-layout -> A-layout) via LDS ----
        __syncthreads();
#pragma unroll
        for (int r = 0; r < 8; ++r) {
            P[kg * 8 + r][ln]      = p0[r];
            P[kg * 8 + r][16 + ln] = p1[r];
        }
        __syncthreads();
        union { v16bf v; __bf16 e[16]; } pa;
#pragma unroll
        for (int e = 0; e < 8; ++e) {
            pa.e[e]     = (__bf16)P[ln][kg * 16 + e];
            pa.e[8 + e] = (__bf16)P[ln][kg * 16 + 8 + e];
        }

        // ---- O += P @ V (V transposed -> contiguous B fragments) ----
        const __bf16* vb = Vh + (size_t)ln * S_LEN + kb + kg * 16;
        o0 = wmma_bf16(pa.v, load_frag2(vb + 0  * 16 * S_LEN, vb + 0  * 16 * S_LEN + 8), o0);
        o1 = wmma_bf16(pa.v, load_frag2(vb + 1  * 16 * S_LEN, vb + 1  * 16 * S_LEN + 8), o1);
        o2 = wmma_bf16(pa.v, load_frag2(vb + 2  * 16 * S_LEN, vb + 2  * 16 * S_LEN + 8), o2);
        o3 = wmma_bf16(pa.v, load_frag2(vb + 3  * 16 * S_LEN, vb + 3  * 16 * S_LEN + 8), o3);
    }

    // ---- epilogue: normalize and write (S, H*DH) ----
#pragma unroll
    for (int r = 0; r < 8; ++r) {
        const int row = qbase + kg * 8 + r;
        const float inv = 1.0f / lrow[r];
        float* op = out + (size_t)row * HIDN + h * DHD;
        op[0  + ln] = o0[r] * inv;
        op[16 + ln] = o1[r] * inv;
        op[32 + ln] = o2[r] * inv;
        op[48 + ln] = o3[r] * inv;
    }
}

// ---------------------------------------------------------------------------
extern "C" void kernel_launch(void* const* d_in, const int* in_sizes, int n_in,
                              void* d_out, int out_size, void* d_ws, size_t ws_size,
                              hipStream_t stream) {
    const float* x    = (const float*)d_in[0];
    const float* Wq   = (const float*)d_in[1];
    const float* bq   = (const float*)d_in[2];
    const float* Wk   = (const float*)d_in[3];
    const float* bk   = (const float*)d_in[4];
    const float* Wv   = (const float*)d_in[5];
    const float* bv   = (const float*)d_in[6];
    const float* Wtq  = (const float*)d_in[7];
    const float* btq  = (const float*)d_in[8];
    const float* Wtk  = (const float*)d_in[9];
    const float* btk  = (const float*)d_in[10];
    const float* Wbil = (const float*)d_in[11];
    const float* tbia = (const float*)d_in[12];
    float* out = (float*)d_out;

    char* wsb = (char*)d_ws;
    size_t off = 0;
    auto take = [&](size_t bytes) -> char* {
        char* p = wsb + off;
        off = (off + bytes + 255) & ~(size_t)255;
        return p;
    };
    const int NTD = NH * TDD;                       // 512
    __bf16* Xbf  = (__bf16*)take((size_t)S_LEN * HIDN * 2);
    __bf16* WqT  = (__bf16*)take((size_t)HIDN * HIDN * 2);
    __bf16* WkT  = (__bf16*)take((size_t)HIDN * HIDN * 2);
    __bf16* WvT  = (__bf16*)take((size_t)HIDN * HIDN * 2);
    __bf16* WtqT = (__bf16*)take((size_t)HIDN * NTD * 2);
    __bf16* WtkT = (__bf16*)take((size_t)HIDN * NTD * 2);
    float*  Qf   = (float*)take((size_t)S_LEN * HIDN * 4);
    float*  Kf   = (float*)take((size_t)S_LEN * HIDN * 4);
    float*  Vf   = (float*)take((size_t)S_LEN * HIDN * 4);
    float*  QTf  = (float*)take((size_t)S_LEN * NTD * 4);
    float*  KTf  = (float*)take((size_t)S_LEN * NTD * 4);
    __bf16* Qr   = (__bf16*)take((size_t)NH * S_LEN * DHD * 2);
    __bf16* Kr   = (__bf16*)take((size_t)NH * S_LEN * DHD * 2);
    __bf16* Vt   = (__bf16*)take((size_t)NH * S_LEN * DHD * 2);
    __bf16* QTW  = (__bf16*)take((size_t)NH * S_LEN * TDD * 2);
    __bf16* KTo  = (__bf16*)take((size_t)NH * S_LEN * TDD * 2);

    // 1) convert activations + transpose/convert weights to bf16
    int nx = S_LEN * HIDN;
    cvt_f32_bf16<<<(nx + 255) / 256, 256, 0, stream>>>(x, Xbf, nx);
    transpose_cvt<<<(HIDN * HIDN + 255) / 256, 256, 0, stream>>>(Wq,  WqT,  HIDN, HIDN);
    transpose_cvt<<<(HIDN * HIDN + 255) / 256, 256, 0, stream>>>(Wk,  WkT,  HIDN, HIDN);
    transpose_cvt<<<(HIDN * HIDN + 255) / 256, 256, 0, stream>>>(Wv,  WvT,  HIDN, HIDN);
    transpose_cvt<<<(HIDN * NTD  + 255) / 256, 256, 0, stream>>>(Wtq, WtqT, HIDN, NTD);
    transpose_cvt<<<(HIDN * NTD  + 255) / 256, 256, 0, stream>>>(Wtk, WtkT, HIDN, NTD);

    // 2) WMMA projection GEMMs
    dim3 g1(HIDN / 64, S_LEN / 32);
    gemm_bf16<<<g1, 128, 0, stream>>>(Xbf, WqT, bq, Qf, S_LEN, HIDN, HIDN);
    gemm_bf16<<<g1, 128, 0, stream>>>(Xbf, WkT, bk, Kf, S_LEN, HIDN, HIDN);
    gemm_bf16<<<g1, 128, 0, stream>>>(Xbf, WvT, bv, Vf, S_LEN, HIDN, HIDN);
    dim3 g2(NTD / 64, S_LEN / 32);
    gemm_bf16<<<g2, 128, 0, stream>>>(Xbf, WtqT, btq, QTf, S_LEN, NTD, HIDN);
    gemm_bf16<<<g2, 128, 0, stream>>>(Xbf, WtkT, btk, KTf, S_LEN, NTD, HIDN);

    // 3) RoPE + layouts + bilinear
    prep_kernel<<<dim3(S_LEN, NH), 64, 0, stream>>>(Qf, Kf, Vf, QTf, KTf, Wbil,
                                                    Qr, Kr, Vt, QTW, KTo);

    // 4) flash attention with WMMA
    attn_kernel<<<dim3(S_LEN / 64, NH), 128, 0, stream>>>(Qr, Kr, Vt, QTW, KTo,
                                                          tbia, out);
}
